// ImprovedActorCriticNetwork_49752901157010
// MI455X (gfx1250) — compile-verified, matrix operand order checked by reading, not run on hardware
//
#include <hip/hip_runtime.h>
#include <hip/hip_bf16.h>

typedef __attribute__((ext_vector_type(16))) __bf16 v16bf;
typedef __attribute__((ext_vector_type(8)))  __bf16 v8bf;
typedef __attribute__((ext_vector_type(8)))  float  v8f;

#define NNODES 32768
#define NEDGES 131072
#define DDIM   512
#define QKVS_W 2048   // q|k|v|s concatenated
#define NHEAD  8
#define CCH    64

// ---------------- helpers: order-preserving float<->uint for atomicMax ----------------
__device__ __forceinline__ unsigned enc_key(float f) {
  unsigned u = __float_as_uint(f);
  return (u & 0x80000000u) ? ~u : (u | 0x80000000u);
}
__device__ __forceinline__ float dec_key(unsigned k) {
  unsigned u = (k & 0x80000000u) ? (k & 0x7fffffffu) : ~k;
  return __uint_as_float(u);
}

// async 16B/lane copy global -> LDS, GVS addressing (SGPR base + 32-bit VGPR offset)
__device__ __forceinline__ void async_copy16s(unsigned lds_off, unsigned voff,
                                              unsigned long long saddr) {
  asm volatile("global_load_async_to_lds_b128 %0, %1, %2"
               :: "v"(lds_off), "v"(voff), "s"(saddr)
               : "memory");
}
__device__ __forceinline__ void wait_async0() {
  asm volatile("s_wait_asynccnt 0x0" ::: "memory");
}

// ---------------- build x0 [N,6] ----------------
__global__ void build_x0_kernel(const float* __restrict__ mc, const int* __restrict__ amask,
                                const float* __restrict__ speeds, const float* __restrict__ cost,
                                const float* __restrict__ arr, float* __restrict__ x0) {
  int n = blockIdx.x * 256 + threadIdx.x;      // N threads
  int m = n >> 3, u = n & 7;
  float* r = x0 + (size_t)n * 6;
  r[0] = mc[m * 2 + 0];
  r[1] = mc[m * 2 + 1];
  r[2] = (float)amask[m];
  r[3] = speeds[u];
  r[4] = cost[(size_t)u * 4096 + m];
  r[5] = arr[(size_t)u * 4096 + m];
}

// ---------------- layer-0 linears (K=6, scalar) ----------------
__global__ void qkvs0_kernel(const float* __restrict__ x0,
                             const float* __restrict__ Wq, const float* __restrict__ bq,
                             const float* __restrict__ Wk, const float* __restrict__ bk,
                             const float* __restrict__ Wv, const float* __restrict__ bv,
                             const float* __restrict__ Ws, const float* __restrict__ bs,
                             float* __restrict__ qkvs) {
  size_t idx = (size_t)blockIdx.x * 256 + threadIdx.x;   // N*2048
  int n = (int)(idx >> 11);
  int j = (int)(idx & 2047);
  int g = j >> 9, col = j & 511;
  const float* W; const float* b;
  switch (g) {
    case 0: W = Wq; b = bq; break;
    case 1: W = Wk; b = bk; break;
    case 2: W = Wv; b = bv; break;
    default: W = Ws; b = bs; break;
  }
  const float* xr = x0 + (size_t)n * 6;
  float acc = b[col];
  #pragma unroll
  for (int i = 0; i < 6; ++i) acc += xr[i] * W[i * 512 + col];
  qkvs[idx] = acc;
}

// ---------------- weight pre-swizzle into WMMA B-fragment layout ----------------
// Wsw layout: [layer][ct(128)][kt(16)][lane(32)][16 bf16]  (one contiguous frag per lane)
__global__ void convert_w_kernel(const float* __restrict__ Wq, const float* __restrict__ Wk,
                                 const float* __restrict__ Wv, const float* __restrict__ Ws,
                                 __bf16* __restrict__ Wsw) {
  size_t idx = (size_t)blockIdx.x * 256 + threadIdx.x;   // 3*128*16*32*16 = 3,145,728
  int i    = (int)(idx & 15);
  int lane = (int)((idx >> 4) & 31);
  int kt   = (int)((idx >> 9) & 15);
  int ct   = (int)((idx >> 13) & 127);
  int l    = (int)(idx >> 20);
  int p = i >> 1, odd = i & 1;
  int hi = lane >> 4, lr = lane & 15;
  int k = kt * 32 + ((p < 4) ? (hi * 8 + 2 * p) : (16 + hi * 8 + 2 * (p - 4))) + odd;
  int j = ct * 16 + lr;
  int g = j >> 9, col = j & 511;
  const float* W = (g == 0) ? Wq : (g == 1) ? Wk : (g == 2) ? Wv : Ws;
  Wsw[idx] = (__bf16)W[((size_t)l * 512 + k) * 512 + col];
}

__global__ void convert_b_kernel(const float* __restrict__ bq, const float* __restrict__ bk,
                                 const float* __restrict__ bv, const float* __restrict__ bs,
                                 float* __restrict__ bcat) {
  int idx = blockIdx.x * 256 + threadIdx.x;              // 3*2048
  int l = idx >> 11, j = idx & 2047;
  int g = j >> 9, col = j & 511;
  const float* b = (g == 0) ? bq : (g == 1) ? bk : (g == 2) ? bv : bs;
  bcat[idx] = b[l * 512 + col];
}

// ---------------- fused QKVS GEMM: [N,512](bf16) x [512,2048](bf16) + bias -> f32 ----------
// Block (8 waves) -> 128x128 output tile. K stepped by 32, double-buffered LDS staging
// via global_load_async_to_lds_b128 (SADDR form, ASYNCcnt). Each wave: 2 row tiles x
// 4 col tiles -> 8 v_wmma_f32_16x16x32_bf16 per K-step from contiguous 32B LDS frags.
#define TILE_BYTES 16384                 // 8KB A + 8KB B per buffer
__global__ void __launch_bounds__(256) gemm_qkvs_wmma(const __bf16* __restrict__ X,
                                                      const __bf16* __restrict__ Wsw,
                                                      const float* __restrict__ bias,
                                                      float* __restrict__ out) {
  __shared__ __align__(16) char smem[2 * TILE_BYTES];

  const int t      = threadIdx.x;
  const int wave   = t >> 5;
  const int rtg    = wave >> 1;          // 0..3 : row-tile pair {2*rtg, 2*rtg+1}
  const int ctg    = wave & 1;           // 0..1 : col tiles {4*ctg .. 4*ctg+3}
  const int lane   = t & 31;
  const int hi     = lane >> 4;
  const int lr     = lane & 15;
  const int rowblk = blockIdx.x >> 4;    // 256 row blocks of 128 rows
  const int colblk = blockIdx.x & 15;    // 16 col blocks of 128 cols
  const int n_base = colblk * 128;

  const unsigned lds0 = (unsigned)(size_t)(void*)smem;   // LDS offset of smem
  const unsigned long long saddrA =
      (unsigned long long)(size_t)X + (size_t)rowblk * 128 * DDIM * 2;
  const unsigned long long saddrB =
      (unsigned long long)(size_t)Wsw + (size_t)colblk * 8 * 16 * 1024;

  // per-thread copy offsets (kt-invariant), 2 x 16B chunks each for A and B
  unsigned gA[2], lA[2], gB[2], lB[2];
  #pragma unroll
  for (int rr = 0; rr < 2; ++rr) {
    int c = t + rr * 256;                // 512 chunks of 16B per tile
    int row = c >> 2, kc = c & 3;        // A: (row, kc) -> frag slot (rt, hi=kc&1, lr)
    gA[rr] = (unsigned)(row * 1024 + kc * 16);
    lA[rr] = lds0 +
        (unsigned)(((row >> 4) * 32 + (kc & 1) * 16 + (row & 15)) * 32 + (kc >> 1) * 16);
    gB[rr] = (unsigned)((c >> 6) * 16384 + (c & 63) * 16);   // B frag layout verbatim
    lB[rr] = lds0 + 8192 + (unsigned)c * 16;
  }

  // stage one K-step (16KB) into buffer `buf`
  auto issue_step = [&](int kt, int buf) {
    const unsigned bo = (unsigned)buf * TILE_BYTES;
    const unsigned koA = (unsigned)kt * 64;     // A row stride 1024B, K advance 64B
    const unsigned koB = (unsigned)kt * 1024;   // B kt chunk = 1KB within 16KB ct block
    #pragma unroll
    for (int rr = 0; rr < 2; ++rr) {
      async_copy16s(lA[rr] + bo, gA[rr] + koA, saddrA);
      async_copy16s(lB[rr] + bo, gB[rr] + koB, saddrB);
    }
  };

  v8f acc[2][4];
  #pragma unroll
  for (int j = 0; j < 4; ++j) {
    float b = bias[n_base + (ctg * 4 + j) * 16 + lr];
    #pragma unroll
    for (int r = 0; r < 8; ++r) { acc[0][j][r] = b; acc[1][j][r] = b; }
  }

  issue_step(0, 0);
  wait_async0();
  __syncthreads();

  for (int kt = 0; kt < 16; ++kt) {
    const int cur = kt & 1;
    if (kt < 15) issue_step(kt + 1, cur ^ 1);   // overlap DMA of next step with WMMAs

    const char* bufA = smem + cur * TILE_BYTES;
    const char* bufB = bufA + 8192;
    v16bf a0 = *(const v16bf*)(bufA + ((2 * rtg + 0) * 32 + lane) * 32);
    v16bf a1 = *(const v16bf*)(bufA + ((2 * rtg + 1) * 32 + lane) * 32);
    #pragma unroll
    for (int j = 0; j < 4; ++j) {
      v16bf bb = *(const v16bf*)(bufB + ((ctg * 4 + j) * 32 + lane) * 32);
      acc[0][j] = __builtin_amdgcn_wmma_f32_16x16x32_bf16(
          false, a0, false, bb, (short)0, acc[0][j], false, false);
      acc[1][j] = __builtin_amdgcn_wmma_f32_16x16x32_bf16(
          false, a1, false, bb, (short)0, acc[1][j], false, false);
    }

    wait_async0();       // own async copies for next buffer landed
    __syncthreads();     // everyone done reading cur + sees next buffer
  }

  #pragma unroll
  for (int i = 0; i < 2; ++i) {
    #pragma unroll
    for (int j = 0; j < 4; ++j) {
      #pragma unroll
      for (int r = 0; r < 8; ++r) {
        int m = rowblk * 128 + (2 * rtg + i) * 16 + r + 8 * hi;
        out[(size_t)m * QKVS_W + n_base + (ctg * 4 + j) * 16 + lr] = acc[i][j][r];
      }
    }
  }
}

// ---------------- attention: reset / alpha+segmax / exp+denom / aggregate / finalize ----------------
__global__ void reset_attn_kernel(float* __restrict__ agg, unsigned* __restrict__ amax,
                                  float* __restrict__ denom) {
  size_t idx = (size_t)blockIdx.x * 256 + threadIdx.x;   // N*512
  agg[idx] = 0.f;
  if (idx < (size_t)NNODES * NHEAD) {
    amax[idx] = enc_key(-3.0e38f);
    denom[idx] = 0.f;
  }
}

__global__ void attn_alpha_kernel(const float* __restrict__ qkvs, const int* __restrict__ ei,
                                  float* __restrict__ alpha, unsigned* __restrict__ amax) {
  int idx = blockIdx.x * 256 + threadIdx.x;              // E*H
  int e = idx >> 3, h = idx & 7;
  int src = ei[e], dst = ei[NEDGES + e];
  const float* q = qkvs + (size_t)dst * QKVS_W + h * CCH;          // q block
  const float* k = qkvs + (size_t)src * QKVS_W + 512 + h * CCH;    // k block
  float acc = 0.f;
  #pragma unroll 8
  for (int c = 0; c < CCH; ++c) acc += q[c] * k[c];
  acc *= 0.125f;                                          // 1/sqrt(64)
  alpha[idx] = acc;
  atomicMax(&amax[dst * NHEAD + h], enc_key(acc));
}

__global__ void attn_ex_kernel(const int* __restrict__ ei, float* __restrict__ alpha,
                               const unsigned* __restrict__ amax, float* __restrict__ denom) {
  int idx = blockIdx.x * 256 + threadIdx.x;              // E*H
  int e = idx >> 3, h = idx & 7;
  int dst = ei[NEDGES + e];
  float m = dec_key(amax[dst * NHEAD + h]);
  float ex = __expf(alpha[idx] - m);
  alpha[idx] = ex;                                        // reuse buffer for ex
  atomicAdd(&denom[dst * NHEAD + h], ex);
}

__global__ void attn_agg_kernel(const float* __restrict__ qkvs, const int* __restrict__ ei,
                                const float* __restrict__ ex, float* __restrict__ agg) {
  size_t idx = (size_t)blockIdx.x * 256 + threadIdx.x;   // E*512
  int e = (int)(idx >> 9);
  int d = (int)(idx & 511);
  int h = d >> 6;
  int src = ei[e], dst = ei[NEDGES + e];
  float w = ex[e * NHEAD + h];
  float v = qkvs[(size_t)src * QKVS_W + 1024 + d];        // v block
  atomicAdd(&agg[(size_t)dst * DDIM + d], w * v);
}

__global__ void finalize_kernel(const float* __restrict__ agg, const float* __restrict__ denom,
                                const float* __restrict__ qkvs, float* __restrict__ x_f32,
                                __bf16* __restrict__ x_bf16) {
  size_t idx = (size_t)blockIdx.x * 256 + threadIdx.x;   // N*512
  int n = (int)(idx >> 9);
  int d = (int)(idx & 511);
  int h = d >> 6;
  float o = agg[idx] / (denom[n * NHEAD + h] + 1e-16f)
          + qkvs[(size_t)n * QKVS_W + 1536 + d];          // root skip
  o = fmaxf(o, 0.f);                                      // relu
  x_f32[idx] = o;
  x_bf16[idx] = (__bf16)o;
}

// ---------------- mean pool ----------------
__global__ void zero_pooled_kernel(float* __restrict__ pooled) {
  pooled[threadIdx.x] = 0.f;                              // 512 threads
}

__global__ void pool_kernel(const float* __restrict__ x, float* __restrict__ pooled) {
  int b = blockIdx.x;                                     // 1024 blocks, 32 rows each
  int t = threadIdx.x;                                    // 256 threads
  #pragma unroll
  for (int cp = 0; cp < 2; ++cp) {
    int c = t + cp * 256;
    float acc = 0.f;
    for (int r = 0; r < 32; ++r) acc += x[(size_t)(b * 32 + r) * DDIM + c];
    atomicAdd(&pooled[c], acc);
  }
}

// ---------------- heads: fc -> actor softmax + critic ----------------
__global__ void heads_kernel(const float* __restrict__ pooled, const float* __restrict__ uavs,
                             const float* __restrict__ speeds,
                             const float* __restrict__ fcW, const float* __restrict__ fcb,
                             const float* __restrict__ aW1, const float* __restrict__ ab1,
                             const float* __restrict__ aW2, const float* __restrict__ ab2,
                             const float* __restrict__ cW1, const float* __restrict__ cb1,
                             const float* __restrict__ cW2, const float* __restrict__ cb2,
                             float* __restrict__ out) {
  const int u = blockIdx.x;                               // U=8 blocks
  const int t = threadIdx.x;                              // 256 threads
  __shared__ float s_comb[68];
  __shared__ float s_hidA[128];
  __shared__ float s_hidC[128];
  __shared__ float s_red[256];

  if (t < 64) {
    float acc = fcb[t];
    const float invN = 1.0f / (float)NNODES;
    for (int k = 0; k < DDIM; ++k) acc += (pooled[k] * invN) * fcW[k * 64 + t];
    s_comb[2 + t] = acc;
  }
  if (t == 64) {
    s_comb[0] = uavs[u * 2 + 0];
    s_comb[1] = uavs[u * 2 + 1];
    s_comb[66] = speeds[u];
  }
  __syncthreads();

  if (t < 128) {
    float a = ab1[t], c = cb1[t];
    for (int i = 0; i < 67; ++i) {
      float x = s_comb[i];
      a += x * aW1[i * 128 + t];
      c += x * cW1[i * 128 + t];
    }
    s_hidA[t] = fmaxf(a, 0.f);
    s_hidC[t] = fmaxf(c, 0.f);
  }
  __syncthreads();

  float* probs = out + (size_t)u * 4096;
  float lmax = -3.4e38f;
  for (int m = t; m < 4096; m += 256) {
    float l = ab2[m];
    for (int j = 0; j < 128; ++j) l += s_hidA[j] * aW2[j * 4096 + m];
    probs[m] = l;
    lmax = fmaxf(lmax, l);
  }
  s_red[t] = lmax; __syncthreads();
  for (int s = 128; s > 0; s >>= 1) { if (t < s) s_red[t] = fmaxf(s_red[t], s_red[t + s]); __syncthreads(); }
  lmax = s_red[0]; __syncthreads();

  float lsum = 0.f;
  for (int m = t; m < 4096; m += 256) {
    float e = __expf(probs[m] - lmax);
    probs[m] = e;
    lsum += e;
  }
  s_red[t] = lsum; __syncthreads();
  for (int s = 128; s > 0; s >>= 1) { if (t < s) s_red[t] += s_red[t + s]; __syncthreads(); }
  lsum = s_red[0]; __syncthreads();
  float inv = 1.0f / lsum;
  for (int m = t; m < 4096; m += 256) probs[m] *= inv;

  float cv = (t < 128) ? s_hidC[t] * cW2[t] : 0.f;
  s_red[t] = cv; __syncthreads();
  for (int s = 128; s > 0; s >>= 1) { if (t < s) s_red[t] += s_red[t + s]; __syncthreads(); }
  if (t == 0) out[8 * 4096 + u] = s_red[0] + cb2[0];
}

// ---------------- launch ----------------
extern "C" void kernel_launch(void* const* d_in, const int* in_sizes, int n_in,
                              void* d_out, int out_size, void* d_ws, size_t ws_size,
                              hipStream_t stream) {
  (void)in_sizes; (void)n_in; (void)out_size; (void)ws_size;
  const float* mission_coords = (const float*)d_in[0];
  const float* uavs_info      = (const float*)d_in[1];
  const float* speeds         = (const float*)d_in[2];
  const float* cost_matrix    = (const float*)d_in[3];
  const float* arrival_times  = (const float*)d_in[4];
  const int*   edge_index     = (const int*)d_in[5];
  const int*   action_mask    = (const int*)d_in[7];
  const float* Wq0 = (const float*)d_in[8];  const float* bq0 = (const float*)d_in[9];
  const float* Wk0 = (const float*)d_in[10]; const float* bk0 = (const float*)d_in[11];
  const float* Wv0 = (const float*)d_in[12]; const float* bv0 = (const float*)d_in[13];
  const float* Ws0 = (const float*)d_in[14]; const float* bs0 = (const float*)d_in[15];
  const float* Wq = (const float*)d_in[16]; const float* bq = (const float*)d_in[17];
  const float* Wk = (const float*)d_in[18]; const float* bk = (const float*)d_in[19];
  const float* Wv = (const float*)d_in[20]; const float* bv = (const float*)d_in[21];
  const float* Ws = (const float*)d_in[22]; const float* bs = (const float*)d_in[23];
  const float* fcW = (const float*)d_in[24]; const float* fcb = (const float*)d_in[25];
  const float* aW1 = (const float*)d_in[26]; const float* ab1 = (const float*)d_in[27];
  const float* aW2 = (const float*)d_in[28]; const float* ab2 = (const float*)d_in[29];
  const float* cW1 = (const float*)d_in[30]; const float* cb1 = (const float*)d_in[31];
  const float* cW2 = (const float*)d_in[32]; const float* cb2 = (const float*)d_in[33];
  float* out = (float*)d_out;

  // workspace carve (offsets in bytes, all 256-aligned)
  char* ws = (char*)d_ws;
  float*    x_f32  = (float*)   (ws + 0);           // N*D f32      = 67,108,864 B
  __bf16*   x_bf16 = (__bf16*)  (ws + 67108864);    // N*D bf16     = 33,554,432 B
  __bf16*   Wsw    = (__bf16*)  (ws + 100663296);   // 3*512*2048   =  6,291,456 B
  float*    bcat   = (float*)   (ws + 106954752);   // 3*2048 f32   =     24,576 B
  float*    qkvs   = (float*)   (ws + 106979328);   // N*2048 f32   = 268,435,456 B
  float*    alpha  = (float*)   (ws + 375414784);   // E*H f32      =  4,194,304 B
  unsigned* amax   = (unsigned*)(ws + 379609088);   // N*H u32      =  1,048,576 B
  float*    denom  = (float*)   (ws + 380657664);   // N*H f32      =  1,048,576 B
  float*    agg    = (float*)   (ws + 381706240);   // N*D f32      = 67,108,864 B
  float*    pooled = (float*)   (ws + 448815104);   // D f32        =      2,048 B

  // feature build + layer-0 linears + weight conversion
  build_x0_kernel<<<NNODES / 256, 256, 0, stream>>>(mission_coords, action_mask, speeds,
                                                    cost_matrix, arrival_times, x_f32);
  qkvs0_kernel<<<(NNODES * (size_t)QKVS_W) / 256, 256, 0, stream>>>(
      x_f32, Wq0, bq0, Wk0, bk0, Wv0, bv0, Ws0, bs0, qkvs);
  convert_w_kernel<<<(3u * 512 * 2048) / 256, 256, 0, stream>>>(Wq, Wk, Wv, Ws, Wsw);
  convert_b_kernel<<<(3 * 2048) / 256, 256, 0, stream>>>(bq, bk, bv, bs, bcat);

  const int gridND   = (NNODES * DDIM) / 256;          // 65536
  const int gridEH   = (NEDGES * NHEAD) / 256;         // 4096
  const int gridED   = (int)(((size_t)NEDGES * DDIM) / 256);  // 262144

  for (int layer = 0; layer < 4; ++layer) {
    if (layer > 0) {
      gemm_qkvs_wmma<<<4096, 256, 0, stream>>>(
          x_bf16, Wsw + (size_t)(layer - 1) * 512 * 2048, bcat + (layer - 1) * 2048, qkvs);
    }
    reset_attn_kernel<<<gridND, 256, 0, stream>>>(agg, amax, denom);
    attn_alpha_kernel<<<gridEH, 256, 0, stream>>>(qkvs, edge_index, alpha, amax);
    attn_ex_kernel<<<gridEH, 256, 0, stream>>>(edge_index, alpha, amax, denom);
    attn_agg_kernel<<<gridED, 256, 0, stream>>>(qkvs, edge_index, alpha, agg);
    finalize_kernel<<<gridND, 256, 0, stream>>>(agg, denom, qkvs, x_f32, x_bf16);
  }

  zero_pooled_kernel<<<1, 512, 0, stream>>>(pooled);
  pool_kernel<<<NNODES / 32, 256, 0, stream>>>(x_f32, pooled);
  heads_kernel<<<8, 256, 0, stream>>>(pooled, uavs_info, speeds, fcW, fcb,
                                      aW1, ab1, aW2, ab2, cW1, cb1, cW2, cb2, out);
}
